// Qwen3Attention_89481348645361
// MI455X (gfx1250) — compile-verified, matrix operand order checked by reading, not run on hardware
//
#include <hip/hip_runtime.h>
#include <hip/hip_bf16.h>

// ---------------------------------------------------------------------------
// Qwen3 attention layer for MI455X (gfx1250), bf16 WMMA pipeline with
// async global->LDS double-buffered GEMM.
// B=2, S=2048, HID=2048, NH=16, NKV=4, HD=128, theta=10000.
// ---------------------------------------------------------------------------

typedef __attribute__((ext_vector_type(8)))  __bf16 v8bf;
typedef __attribute__((ext_vector_type(16))) __bf16 v16bf;
typedef __attribute__((ext_vector_type(8)))  float  v8f;

constexpr int B_   = 2;
constexpr int S_   = 2048;
constexpr int HID_ = 2048;
constexpr int NH_  = 16;
constexpr int NKV_ = 4;
constexpr int HD_  = 128;
constexpr int NT_  = B_ * S_;          // 4096 tokens
constexpr int NQ_  = NH_ * HD_;        // 2048
constexpr int NKVD_ = NKV_ * HD_;      // 512

// ---------------------------------------------------------------------------
// Fragment loader: CDNA5 16-bit A/B layout for 16x16x32 WMMA.
// lane L holds matrix row (A) / matrix column (B) = row0 + (L & 15).
// Within that row: 8 contiguous bf16 at k0 + (L>>4)*8 -> frag[0..7]
//                  8 contiguous bf16 at k0 + 16 + (L>>4)*8 -> frag[8..15]
// Works for global and LDS (generic) pointers.
// ---------------------------------------------------------------------------
__device__ __forceinline__ v16bf load_frag(const __bf16* __restrict__ base,
                                           int ld, int row0, int k0) {
  const int lane = threadIdx.x & 31;
  const int row  = row0 + (lane & 15);
  const int koff = (lane >> 4) << 3;
  const __bf16* p = base + (size_t)row * (size_t)ld + (size_t)(k0 + koff);
  v8bf lo = *(const v8bf*)(p);
  v8bf hi = *(const v8bf*)(p + 16);
  v16bf a;
#pragma unroll
  for (int i = 0; i < 8; ++i) { a[i] = lo[i]; a[i + 8] = hi[i]; }
  return a;
}

__device__ __forceinline__ v8f wmma_bf16(v16bf a, v16bf b, v8f c) {
  return __builtin_amdgcn_wmma_f32_16x16x32_bf16(
      /*neg_a=*/false, a, /*neg_b=*/false, b,
      /*c_mod=*/(short)0, c, /*reuse_a=*/false, /*reuse_b=*/false);
}

// Async global->LDS copy of one 16-byte chunk (per-lane addressing).
// Tracked by ASYNCcnt; completion enforced with s_wait_asynccnt.
__device__ __forceinline__ void async_copy_b128(const __bf16* g, __bf16* l) {
  unsigned lds_off = (unsigned)(unsigned long long)l;   // aperture low 32 bits
  unsigned long long ga = (unsigned long long)g;
  asm volatile("global_load_async_to_lds_b128 %0, %1, off"
               :: "v"(lds_off), "v"(ga) : "memory");
}

// ---------------------------------------------------------------------------
// f32 -> bf16 cast
// ---------------------------------------------------------------------------
__global__ void cast_bf16_kernel(const float* __restrict__ src,
                                 __bf16* __restrict__ dst, int n) {
  int i = blockIdx.x * blockDim.x + threadIdx.x;
  if (i < n) dst[i] = (__bf16)src[i];
}

// ---------------------------------------------------------------------------
// GEMM: C[M,N] = A[M,K] (bf16) x W[N,K]^T (bf16) + bias (f32), C f32.
// Block 256 threads = 8 waves (4 along M x 2 along N); block tile 128x128;
// wave tile 32x64 (2 A-frags x 4 B-frags -> 8 accumulators).
// Global->LDS via async-to-LDS, double buffered; grid = (M/128, N/128).
// ---------------------------------------------------------------------------
constexpr int ROWE = 40;  // padded LDS row length in elements (80 B stride)

__global__ __launch_bounds__(256) void gemm_bf16_kernel(
    const __bf16* __restrict__ A, const __bf16* __restrict__ W,
    const float* __restrict__ bias, float* __restrict__ C,
    int M, int N, int K) {
  __shared__ __bf16 tA[2][128 * ROWE];
  __shared__ __bf16 tB[2][128 * ROWE];

  const int tid  = threadIdx.x;
  const int lane = tid & 31;
  const int w    = tid >> 5;
  const int mblk = blockIdx.x * 128;
  const int nblk = blockIdx.y * 128;
  const int m0w  = (w >> 1) * 32;   // block-local wave M origin
  const int n0w  = (w & 1) * 64;    // block-local wave N origin

  // cooperative stage: thread t owns 32 contiguous bytes of each 128x32 tile
  const int srow = tid >> 1;
  const int scol = (tid & 1) * 16;

  auto issue_tiles = [&](int buf, int k0) {
    const __bf16* gA = A + (size_t)(mblk + srow) * K + k0 + scol;
    __bf16* lA = &tA[buf][srow * ROWE + scol];
    async_copy_b128(gA, lA);
    async_copy_b128(gA + 8, lA + 8);
    const __bf16* gB = W + (size_t)(nblk + srow) * K + k0 + scol;
    __bf16* lB = &tB[buf][srow * ROWE + scol];
    async_copy_b128(gB, lB);
    async_copy_b128(gB + 8, lB + 8);
  };

  const v8f zf = {0.f, 0.f, 0.f, 0.f, 0.f, 0.f, 0.f, 0.f};
  v8f acc[2][4] = {{zf, zf, zf, zf}, {zf, zf, zf, zf}};

  int buf = 0;
  issue_tiles(0, 0);
  for (int k0 = 0; k0 < K; k0 += 32) {
    const bool more = (k0 + 32 < K);
    if (more) {
      issue_tiles(buf ^ 1, k0 + 32);                    // prefetch next buffer
      asm volatile("s_wait_asynccnt 4" ::: "memory");   // current buffer done
    } else {
      asm volatile("s_wait_asynccnt 0" ::: "memory");
    }
    __syncthreads();

    v16bf a0 = load_frag(&tA[buf][0], ROWE, m0w, 0);
    v16bf a1 = load_frag(&tA[buf][0], ROWE, m0w + 16, 0);
#pragma unroll
    for (int j = 0; j < 4; ++j) {
      v16bf b = load_frag(&tB[buf][0], ROWE, n0w + 16 * j, 0);
      acc[0][j] = wmma_bf16(a0, b, acc[0][j]);
      acc[1][j] = wmma_bf16(a1, b, acc[1][j]);
    }
    __syncthreads();
    buf ^= 1;
  }

  const int rowoff = (lane >> 4) << 3;
  const int colL   = lane & 15;
#pragma unroll
  for (int i = 0; i < 2; ++i) {
#pragma unroll
    for (int j = 0; j < 4; ++j) {
#pragma unroll
      for (int r = 0; r < 8; ++r) {
        int row = mblk + m0w + 16 * i + r + rowoff;
        int col = nblk + n0w + 16 * j + colL;
        float v = acc[i][j][r];
        if (bias) v += bias[col];
        C[(size_t)row * N + col] = v;
      }
    }
  }
}

// ---------------------------------------------------------------------------
// RoPE + repack: X f32 [B*S, nheads*HD] -> Out bf16 [B, nheads, S, HD]
// ---------------------------------------------------------------------------
__global__ void rope_pack_kernel(const float* __restrict__ X,
                                 __bf16* __restrict__ Out, int nheads) {
  int idx = blockIdx.x * blockDim.x + threadIdx.x;
  int total = B_ * nheads * S_ * HD_;
  if (idx >= total) return;
  int d = idx % HD_;
  int s = (idx / HD_) % S_;
  int h = (idx / (HD_ * S_)) % nheads;
  int b = idx / (HD_ * S_ * nheads);

  const int ldx = nheads * HD_;
  size_t rowbase = (size_t)(b * S_ + s) * ldx + h * HD_;
  float x = X[rowbase + d];
  float other = (d < 64) ? -X[rowbase + d + 64] : X[rowbase + d - 64];

  int dr = (d < 64) ? d : d - 64;
  float invf = __powf(10000.0f, -(float)dr * (1.0f / 64.0f));
  float ang  = (float)s * invf;
  float c, sn;
  __sincosf(ang, &sn, &c);
  Out[idx] = (__bf16)(x * c + other * sn);
}

// ---------------------------------------------------------------------------
// V pack (transposed): Vf f32 [B*S, NKV*HD] -> Vt bf16 [B, NKV, HD, S]
// ---------------------------------------------------------------------------
__global__ void vpack_kernel(const float* __restrict__ Vf,
                             __bf16* __restrict__ Vt) {
  int idx = blockIdx.x * blockDim.x + threadIdx.x;
  int total = B_ * NKV_ * HD_ * S_;
  if (idx >= total) return;
  int s  = idx % S_;
  int d  = (idx / S_) % HD_;
  int hk = (idx / (S_ * HD_)) % NKV_;
  int b  = idx / (S_ * HD_ * NKV_);
  Vt[idx] = (__bf16)Vf[(size_t)(b * S_ + s) * NKVD_ + hk * HD_ + d];
}

// ---------------------------------------------------------------------------
// Causal flash attention, GQA (4 q-heads per kv-head).
// grid = (B*NH, S/128), block = 256 (8 waves). Each wave owns a 16-query
// tile and streams keys in chunks of 32 with online softmax.
//   Q  bf16 [B, NH,  S, HD]
//   Kt bf16 [B, NKV, S, HD]
//   Vt bf16 [B, NKV, HD, S]
//   O  bf16 [B*S, NH*HD]
// ---------------------------------------------------------------------------
__global__ void attn_kernel(const __bf16* __restrict__ Q,
                            const __bf16* __restrict__ Kt,
                            const __bf16* __restrict__ Vt,
                            __bf16* __restrict__ O) {
  const int lane = threadIdx.x & 31;
  const int w    = threadIdx.x >> 5;
  const int bh = blockIdx.x;
  const int b  = bh / NH_;
  const int h  = bh % NH_;
  const int hk = h / (NH_ / NKV_);
  const int q0 = blockIdx.y * 128 + w * 16;

  const __bf16* qbase = Q  + (size_t)(b * NH_  + h)  * S_  * HD_;
  const __bf16* kbase = Kt + (size_t)(b * NKV_ + hk) * S_  * HD_;
  const __bf16* vbase = Vt + (size_t)(b * NKV_ + hk) * HD_ * S_;

  __shared__ __bf16 lds_p[8][16 * 32];   // per-wave 16x32 P tile
  __bf16* pbuf = &lds_p[w][0];

  // Q fragments for the whole HD=128 contraction (4 x k32 chunks)
  v16bf qa[4];
#pragma unroll
  for (int kk = 0; kk < 4; ++kk) qa[kk] = load_frag(qbase, HD_, q0, 32 * kk);

  const v8f zf = {0.f, 0.f, 0.f, 0.f, 0.f, 0.f, 0.f, 0.f};
  v8f acc[8] = {zf, zf, zf, zf, zf, zf, zf, zf};
  float m[8], l[8];
#pragma unroll
  for (int r = 0; r < 8; ++r) { m[r] = -1e30f; l[r] = 0.f; }

  const float scale = 0.08838834764831845f;  // 1/sqrt(128)
  const int rowoff = (lane >> 4) << 3;
  const int colL   = lane & 15;

  for (int kb = 0; kb < q0 + 16; kb += 32) {
    // ---- scores: S[16x32] = Q[16x128] x K^T[128x32] ----
    v8f sc[2] = {zf, zf};
#pragma unroll
    for (int kk = 0; kk < 4; ++kk) {
      v16bf b0 = load_frag(kbase, HD_, kb,      32 * kk);
      v16bf b1 = load_frag(kbase, HD_, kb + 16, 32 * kk);
      sc[0] = wmma_bf16(qa[kk], b0, sc[0]);
      sc[1] = wmma_bf16(qa[kk], b1, sc[1]);
    }
    // scale + causal mask (C layout: row = r + 8*(lane>>4), col = lane&15)
#pragma unroll
    for (int t = 0; t < 2; ++t) {
#pragma unroll
      for (int r = 0; r < 8; ++r) {
        int qpos = q0 + r + rowoff;
        int kpos = kb + 16 * t + colL;
        float v = sc[t][r] * scale;
        sc[t][r] = (kpos <= qpos) ? v : -1e30f;
      }
    }
    // ---- online softmax update (row reduction across 16-lane half) ----
#pragma unroll
    for (int r = 0; r < 8; ++r) {
      float rmax = fmaxf(sc[0][r], sc[1][r]);
      rmax = fmaxf(rmax, __shfl_xor(rmax, 1));
      rmax = fmaxf(rmax, __shfl_xor(rmax, 2));
      rmax = fmaxf(rmax, __shfl_xor(rmax, 4));
      rmax = fmaxf(rmax, __shfl_xor(rmax, 8));
      float mn   = fmaxf(m[r], rmax);
      float corr = __expf(m[r] - mn);
      float p0 = __expf(sc[0][r] - mn);
      float p1 = __expf(sc[1][r] - mn);
      sc[0][r] = p0; sc[1][r] = p1;
      float rsum = p0 + p1;
      rsum += __shfl_xor(rsum, 1);
      rsum += __shfl_xor(rsum, 2);
      rsum += __shfl_xor(rsum, 4);
      rsum += __shfl_xor(rsum, 8);
      l[r] = l[r] * corr + rsum;
      m[r] = mn;
#pragma unroll
      for (int t2 = 0; t2 < 8; ++t2) acc[t2][r] *= corr;
    }
    // ---- C-layout -> A-layout via per-wave LDS tile ----
#pragma unroll
    for (int t = 0; t < 2; ++t) {
#pragma unroll
      for (int r = 0; r < 8; ++r) {
        pbuf[(r + rowoff) * 32 + 16 * t + colL] = (__bf16)sc[t][r];
      }
    }
    asm volatile("s_wait_dscnt 0\n\ts_wait_storecnt 0" ::: "memory");
    v16bf pa = load_frag(pbuf, 32, 0, 0);
    // ---- O += P[16x32] x V[32x128] (V stored transposed) ----
#pragma unroll
    for (int t2 = 0; t2 < 8; ++t2) {
      v16bf bv = load_frag(vbase, S_, 16 * t2, kb);
      acc[t2] = wmma_bf16(pa, bv, acc[t2]);
    }
  }

  // epilogue: divide by l, store bf16 [B*S, NH*HD]
#pragma unroll
  for (int t2 = 0; t2 < 8; ++t2) {
#pragma unroll
    for (int r = 0; r < 8; ++r) {
      int qpos = q0 + r + rowoff;
      int d    = 16 * t2 + colL;
      float o  = acc[t2][r] / l[r];
      O[(size_t)(b * S_ + qpos) * NQ_ + h * HD_ + d] = (__bf16)o;
    }
  }
}

// ---------------------------------------------------------------------------
// Host launch
// ---------------------------------------------------------------------------
extern "C" void kernel_launch(void* const* d_in, const int* in_sizes, int n_in,
                              void* d_out, int out_size, void* d_ws, size_t ws_size,
                              hipStream_t stream) {
  const float* hidden = (const float*)d_in[0];
  // d_in[1] = attention_mask (causal; applied analytically, unused)
  const float* Wq = (const float*)d_in[2];
  const float* bq = (const float*)d_in[3];
  const float* Wk = (const float*)d_in[4];
  const float* bk = (const float*)d_in[5];
  const float* Wv = (const float*)d_in[6];
  const float* bv = (const float*)d_in[7];
  const float* Wo = (const float*)d_in[8];
  float* out = (float*)d_out;

  char* ws = (char*)d_ws;
  size_t off = 0;
  auto carve = [&](size_t bytes) -> char* {
    char* p = ws + off;
    off = (off + bytes + 255) & ~(size_t)255;
    return p;
  };
  __bf16* Xbf  = (__bf16*)carve((size_t)NT_ * HID_ * 2);
  __bf16* Wqbf = (__bf16*)carve((size_t)NQ_ * HID_ * 2);
  __bf16* Wkbf = (__bf16*)carve((size_t)NKVD_ * HID_ * 2);
  __bf16* Wvbf = (__bf16*)carve((size_t)NKVD_ * HID_ * 2);
  __bf16* Wobf = (__bf16*)carve((size_t)HID_ * NQ_ * 2);
  float*  Qf   = (float*) carve((size_t)NT_ * NQ_ * 4);
  float*  Kf   = (float*) carve((size_t)NT_ * NKVD_ * 4);
  float*  Vf   = (float*) carve((size_t)NT_ * NKVD_ * 4);
  __bf16* Qbf  = (__bf16*)carve((size_t)NT_ * NQ_ * 2);
  __bf16* Kbf  = (__bf16*)carve((size_t)NT_ * NKVD_ * 2);
  __bf16* Vtbf = (__bf16*)carve((size_t)NT_ * NKVD_ * 2);
  __bf16* Obf  = (__bf16*)carve((size_t)NT_ * NQ_ * 2);

  const int TPB = 256;
  auto blocks = [](int n) { return (n + 255) / 256; };

  // 1) casts to bf16
  cast_bf16_kernel<<<blocks(NT_ * HID_), TPB, 0, stream>>>(hidden, Xbf, NT_ * HID_);
  cast_bf16_kernel<<<blocks(NQ_ * HID_), TPB, 0, stream>>>(Wq, Wqbf, NQ_ * HID_);
  cast_bf16_kernel<<<blocks(NKVD_ * HID_), TPB, 0, stream>>>(Wk, Wkbf, NKVD_ * HID_);
  cast_bf16_kernel<<<blocks(NKVD_ * HID_), TPB, 0, stream>>>(Wv, Wvbf, NKVD_ * HID_);
  cast_bf16_kernel<<<blocks(HID_ * NQ_), TPB, 0, stream>>>(Wo, Wobf, HID_ * NQ_);

  // 2) QKV projections (WMMA GEMMs, async double-buffered LDS)
  gemm_bf16_kernel<<<dim3(NT_ / 128, NQ_ / 128), TPB, 0, stream>>>(
      Xbf, Wqbf, bq, Qf, NT_, NQ_, HID_);
  gemm_bf16_kernel<<<dim3(NT_ / 128, NKVD_ / 128), TPB, 0, stream>>>(
      Xbf, Wkbf, bk, Kf, NT_, NKVD_, HID_);
  gemm_bf16_kernel<<<dim3(NT_ / 128, NKVD_ / 128), TPB, 0, stream>>>(
      Xbf, Wvbf, bv, Vf, NT_, NKVD_, HID_);

  // 3) RoPE + repack
  rope_pack_kernel<<<blocks(B_ * NH_ * S_ * HD_), TPB, 0, stream>>>(Qf, Qbf, NH_);
  rope_pack_kernel<<<blocks(B_ * NKV_ * S_ * HD_), TPB, 0, stream>>>(Kf, Kbf, NKV_);
  vpack_kernel<<<blocks(B_ * NKV_ * HD_ * S_), TPB, 0, stream>>>(Vf, Vtbf);

  // 4) causal flash attention (WMMA)
  attn_kernel<<<dim3(B_ * NH_, S_ / 128), TPB, 0, stream>>>(Qbf, Kbf, Vtbf, Obf);

  // 5) output projection (WMMA GEMM) -> f32 out
  gemm_bf16_kernel<<<dim3(NT_ / 128, HID_ / 128), TPB, 0, stream>>>(
      Obf, Wobf, nullptr, out, NT_, HID_, NQ_);
}
